// EGNN_53523882442972
// MI455X (gfx1250) — matrix operand order repaired and checked by reading, not compiled, not run on hardware
//
#include <hip/hip_runtime.h>
#include <hip/hip_bf16.h>
#include <hip/hip_fp16.h>

// ---------------------------------------------------------------------------
// EGNN layer for MI455X (gfx1250): all dense layers on V_WMMA_F32_16X16X32_F16.
//   b=2, n=4096, k=32, d0=64, d1=32, hid=32, edge_in=161
// Edge GEMM dims (padded): K1=192 (6 kt), N1=352 (22 nt), K2=352 (11 kt),
//   N2=32 (2 nt), K3=32 (1 kt), N3=128 (8 nt), K4=128 (4 kt), N4=32 (2 nt).
// Each edge wave owns one node's full 32-neighbor list (M=32 rows): B-matrix
// fragments stay in VGPRs across two 16-row M sub-tiles, masked m_i sum and
// htype einsum are single-writer stores (no atomics / zero-init).
// Sigmoid/SiLU epilogues use v_rcp_f32 instead of the IEEE division sequence
// so the VALU epilogue co-executes cheaply with the WMMA (TRANS-class) chain.
// ---------------------------------------------------------------------------

typedef __attribute__((ext_vector_type(16))) _Float16 v16h;
typedef __attribute__((ext_vector_type(8)))  _Float16 v8h;
typedef __attribute__((ext_vector_type(8)))  float    v8f;

#define WMMA_F16(a, b, c) \
  __builtin_amdgcn_wmma_f32_16x16x32_f16(false, (a), false, (b), (short)0, (c), false, false)

// Fast sigmoid: v_exp_f32 + v_rcp_f32 (no v_div_scale/div_fmas expansion).
__device__ __forceinline__ float sigmoid_f(float x) {
  return __builtin_amdgcn_rcpf(1.0f + __expf(-x));
}
__device__ __forceinline__ float silu_f(float x) { return x * sigmoid_f(x); }

// A-fragment (16x32 f16) from a row-major f16 LDS tile.
// Lane l: M = l&15 ; hs = l>>4 selects K ranges per the ISA 16-bit A layout:
//   halves 0..7  -> K = kbase + hs*8 + i
//   halves 8..15 -> K = kbase + 16 + hs*8 + i
__device__ __forceinline__ v16h load_a_frag(const _Float16* rowBase, int kbase, int hs) {
  const v8h* p0 = (const v8h*)(rowBase + kbase + hs * 8);
  const v8h* p1 = (const v8h*)(rowBase + kbase + 16 + hs * 8);
  v8h lo = *p0;            // ds_load_b128
  v8h hi = *p1;            // ds_load_b128
  v16h r;
#pragma unroll
  for (int i = 0; i < 8; ++i) { r[i] = lo[i]; r[8 + i] = hi[i]; }
  return r;
}

// B-fragment (32x16 f16) from pre-swizzled global weights (one 32B load).
__device__ __forceinline__ v16h load_b_frag(const _Float16* wf, int NT, int kt, int nt, int lane) {
  return *(const v16h*)(wf + (((size_t)kt * NT + nt) * 32 + (size_t)lane) * 16);
}

// D layout (16x16 f32, 8 VGPRs): lane l, acc[v] -> M = v + 8*(l>>4), N = l&15.
__device__ __forceinline__ void store_act_f16(v8f acc, float bias, _Float16* dst,
                                              int strideH, int nt, int lane, bool doSilu) {
  int col = nt * 16 + (lane & 15);
  int mbase = (lane >> 4) * 8;
#pragma unroll
  for (int v = 0; v < 8; ++v) {
    float x = acc[v] + bias;
    if (doSilu) x = silu_f(x);
    dst[(mbase + v) * strideH + col] = (_Float16)x;
  }
}

// ---------------------------------------------------------------------------
// Swizzle fp32 weights [K,N] into f16 WMMA-B fragment order, zero padded.
__global__ void prep_weights_kernel(const float* s0, const float* s1, const float* s2,
                                    const float* s3, const float* s4, const float* s5,
                                    const float* s6, _Float16* dst) {
  const int Kr[7]  = {161, 322, 32, 128, 96, 128, 64};
  const int Nr[7]  = {322, 32, 128, 32, 128, 64, 32};
  const int KTt[7] = {6, 11, 1, 4, 3, 4, 2};
  const int NTt[7] = {22, 2, 8, 2, 8, 4, 2};
  const int OFF[7] = {0, 67584, 78848, 82944, 87040, 99328, 107520};  // halves

  int wId = blockIdx.y;
  int tile = blockIdx.x;
  int KT = KTt[wId], NT = NTt[wId];
  if (tile >= KT * NT) return;
  const float* srcs[7] = {s0, s1, s2, s3, s4, s5, s6};
  const float* src = srcs[wId];

  int kt = tile / NT, nt = tile % NT;
  int lane = threadIdx.x >> 4;   // 0..31
  int i    = threadIdx.x & 15;   // 0..15
  int kg = kt * 32 + (i >> 3) * 16 + (lane >> 4) * 8 + (i & 7);
  int ng = nt * 16 + (lane & 15);
  float v = (kg < Kr[wId] && ng < Nr[wId]) ? src[(size_t)kg * Nr[wId] + ng] : 0.0f;
  dst[(size_t)OFF[wId] + ((size_t)tile * 32 + lane) * 16 + i] = (_Float16)v;
}

// ---------------------------------------------------------------------------
// Edge pipeline: one wave (32 lanes) per node -> M = 32 edges (full k list).
__global__ void __launch_bounds__(32)
edge_kernel(const float* __restrict__ F0, const float* __restrict__ F1,
            const int* __restrict__ nbrIdxG, const int* __restrict__ nbrMaskG,
            const float* __restrict__ relDistG,
            const _Float16* __restrict__ W1f, const float* __restrict__ b1,
            const _Float16* __restrict__ W2f, const float* __restrict__ b2,
            const _Float16* __restrict__ W3f, const float* __restrict__ b3,
            const _Float16* __restrict__ W4f, const float* __restrict__ b4,
            const float* __restrict__ hnScale, const float* __restrict__ hnBias,
            float* __restrict__ m_acc, float* __restrict__ ht_acc) {
  __shared__ __align__(16) _Float16 Ein[32 * 192];   // edge_in f16
  __shared__ __align__(16) _Float16 ACT1[32 * 352];  // silu(layer1) f16
  __shared__ __align__(16) _Float16 MIJ[32 * 32];    // m_ij f16
  __shared__ __align__(16) _Float16 H1[32 * 128];    // silu(layer3) f16
  __shared__ __align__(16) float    WW[32 * 32];     // layer-4 out (htype weights)
  __shared__ __align__(16) float    RN[32 * 32];     // rel_norm f32
  __shared__ int NB[32];
  __shared__ int MK[32];

  const int lane = threadIdx.x;            // 0..31
  const int t    = blockIdx.x;             // node id, 0..8191
  const int bi   = t >> 12;
  const size_t node = (size_t)t;           // == bi*4096 + ni

  // ---- Phase 0: build edge_in (one lane per edge row) ----------------------
  {
    int r = lane;
    size_t eidx = node * 32 + r;
    int nb = nbrIdxG[eidx];
    NB[r] = nb;
    MK[r] = nbrMaskG[eidx];
    const float* Fi = F0 + node * 64;
    const float* Fj = F0 + ((size_t)bi * 4096 + nb) * 64;
    _Float16* row = &Ein[r * 192];
    for (int j = 0; j < 64; ++j) { row[j] = (_Float16)Fi[j]; row[64 + j] = (_Float16)Fj[j]; }
    const float* F1i = F1 + node * 96;                          // [d1=32][3]
    const float* F1j = F1 + ((size_t)bi * 4096 + nb) * 96;
    for (int d = 0; d < 32; ++d) {
      float dx = F1i[d * 3 + 0] - F1j[d * 3 + 0];
      float dy = F1i[d * 3 + 1] - F1j[d * 3 + 1];
      float dz = F1i[d * 3 + 2] - F1j[d * 3 + 2];
      float nrm = sqrtf(dx * dx + dy * dy + dz * dz);
      row[128 + d] = (_Float16)nrm;
      RN[r * 32 + d] = nrm;
    }
    row[160] = (_Float16)relDistG[eidx];
    for (int j = 161; j < 192; ++j) row[j] = (_Float16)0.0f;
  }
  __syncthreads();

  const int hs = lane >> 4;
  const int lrow = lane & 15;

  // ---- Layer 1: 192 -> 352, silu (B frags reused across 2 M sub-tiles) ----
  for (int nt = 0; nt < 22; ++nt) {
    v16h B[6];
#pragma unroll
    for (int kt = 0; kt < 6; ++kt) B[kt] = load_b_frag(W1f, 22, kt, nt, lane);
    int ng = nt * 16 + lrow;
    float bias = (ng < 322) ? b1[ng] : 0.0f;
#pragma unroll
    for (int mt = 0; mt < 2; ++mt) {
      const _Float16* aRow = &Ein[(mt * 16 + lrow) * 192];
      v8f acc = {};
#pragma unroll
      for (int kt = 0; kt < 6; ++kt)
        acc = WMMA_F16(load_a_frag(aRow, kt * 32, hs), B[kt], acc);
      store_act_f16(acc, bias, &ACT1[mt * 16 * 352], 352, nt, lane, true);
    }
  }
  __syncthreads();

  // ---- Layer 2: 352 -> 32, silu => m_ij -----------------------------------
  for (int nt = 0; nt < 2; ++nt) {
    v16h B[11];
#pragma unroll
    for (int kt = 0; kt < 11; ++kt) B[kt] = load_b_frag(W2f, 2, kt, nt, lane);
    float bias = b2[nt * 16 + lrow];
#pragma unroll
    for (int mt = 0; mt < 2; ++mt) {
      const _Float16* aRow = &ACT1[(mt * 16 + lrow) * 352];
      v8f acc = {};
#pragma unroll
      for (int kt = 0; kt < 11; ++kt)
        acc = WMMA_F16(load_a_frag(aRow, kt * 32, hs), B[kt], acc);
      store_act_f16(acc, bias, &MIJ[mt * 16 * 32], 32, nt, lane, true);
    }
  }
  __syncthreads();

  // ---- masked m_i sum over all 32 neighbors (single writer, no atomics) ---
  {
    int d = lane;
    float s = 0.0f;
    for (int r = 0; r < 32; ++r)
      if (MK[r]) s += (float)MIJ[r * 32 + d];
    m_acc[node * 32 + d] = s;
  }

  // ---- Layer 3: 32 -> 128, silu => h1 -------------------------------------
  for (int nt = 0; nt < 8; ++nt) {
    v16h B0 = load_b_frag(W3f, 8, 0, nt, lane);
    float bias = b3[nt * 16 + lrow];
#pragma unroll
    for (int mt = 0; mt < 2; ++mt) {
      const _Float16* aRow = &MIJ[(mt * 16 + lrow) * 32];
      v8f acc = {};
      acc = WMMA_F16(load_a_frag(aRow, 0, hs), B0, acc);
      store_act_f16(acc, bias, &H1[mt * 16 * 128], 128, nt, lane, true);
    }
  }
  __syncthreads();

  // ---- Layer 4: 128 -> 32, linear => w (f32) ------------------------------
  for (int nt = 0; nt < 2; ++nt) {
    v16h B[4];
#pragma unroll
    for (int kt = 0; kt < 4; ++kt) B[kt] = load_b_frag(W4f, 2, kt, nt, lane);
    int col = nt * 16 + lrow;
    float bias = b4[col];
#pragma unroll
    for (int mt = 0; mt < 2; ++mt) {
      const _Float16* aRow = &H1[(mt * 16 + lrow) * 128];
      v8f acc = {};
#pragma unroll
      for (int kt = 0; kt < 4; ++kt)
        acc = WMMA_F16(load_a_frag(aRow, kt * 32, hs), B[kt], acc);
      int mbase = mt * 16 + hs * 8;
#pragma unroll
      for (int v = 0; v < 8; ++v) WW[(mbase + v) * 32 + col] = acc[v] + bias;
    }
  }
  __syncthreads();

  // ---- htype_update: sum_r normed_rel[r,d,:] * w[r,d] (single writer) -----
  {
    int d = lane;
    float sc = hnScale[d], hb = hnBias[d];
    const float* F1i = F1 + node * 96;
    float f0 = F1i[d * 3 + 0], f1v = F1i[d * 3 + 1], f2 = F1i[d * 3 + 2];
    float a0 = 0.f, a1 = 0.f, a2 = 0.f;
    for (int r = 0; r < 32; ++r) {
      int nb = NB[r];
      float nrm = RN[r * 32 + d];
      float wv  = WW[r * 32 + d];
      float fac = wv * (nrm * sc + hb) * __builtin_amdgcn_rcpf(fmaxf(nrm, 1e-8f));
      size_t jb = ((size_t)bi * 4096 + nb) * 96 + d * 3;
      a0 += fac * (f0 - F1[jb + 0]);
      a1 += fac * (f1v - F1[jb + 1]);
      a2 += fac * (f2 - F1[jb + 2]);
    }
    size_t ob = node * 96 + d * 3;
    ht_acc[ob + 0] = a0;
    ht_acc[ob + 1] = a1;
    ht_acc[ob + 2] = a2;
  }
}

// ---------------------------------------------------------------------------
// Node pipeline: one wave per 16-node tile.
#define NW 2
__global__ void __launch_bounds__(NW * 32)
node_kernel(const float* __restrict__ F0, const float* __restrict__ F1,
            const float* __restrict__ m_acc, const float* __restrict__ ht_acc,
            const _Float16* __restrict__ Wn1f, const float* __restrict__ b_n1,
            const _Float16* __restrict__ Wn2f, const float* __restrict__ b_n2,
            const _Float16* __restrict__ Wgf,  const float* __restrict__ b_g,
            const float* __restrict__ ln_g, const float* __restrict__ ln_b,
            float* __restrict__ out1, float* __restrict__ out2) {
  __shared__ __align__(16) _Float16 NI[NW][16 * 96];    // node_in f16
  __shared__ __align__(16) _Float16 A1[NW][16 * 128];   // silu(act) f16
  __shared__ __align__(16) _Float16 NOh[NW][16 * 64];   // node_out f16
  __shared__ __align__(16) float    GT[NW][16 * 32];    // gate f32

  const int w    = threadIdx.x >> 5;
  const int lane = threadIdx.x & 31;
  const int t    = blockIdx.x * NW + w;                 // 0..511
  const size_t nodeBase = (size_t)t * 16;

  // ---- Phase 0: layernorm(nodes) ++ m_i ------------------------------------
  if (lane < 16) {
    size_t node = nodeBase + lane;
    const float* x = F0 + node * 64;
    float s = 0.f, s2 = 0.f;
    for (int j = 0; j < 64; ++j) { float v = x[j]; s += v; s2 += v * v; }
    float mu = s * (1.0f / 64.0f);
    float var = s2 * (1.0f / 64.0f) - mu * mu;
    float inv = rsqrtf(var + 1e-5f);
    for (int j = 0; j < 64; ++j)
      NI[w][lane * 96 + j] = (_Float16)((x[j] - mu) * inv * ln_g[j] + ln_b[j]);
  } else {
    int r = lane - 16;
    size_t node = nodeBase + r;
    for (int d = 0; d < 32; ++d)
      NI[w][r * 96 + 64 + d] = (_Float16)m_acc[node * 32 + d];
  }
  __syncthreads();

  const int hs = lane >> 4;
  const int lrow = lane & 15;

  // ---- n1: 96 -> 128, silu --------------------------------------------------
  const _Float16* aRowN = &NI[w][lrow * 96];
  for (int nt = 0; nt < 8; ++nt) {
    v8f acc = {};
#pragma unroll
    for (int kt = 0; kt < 3; ++kt) {
      v16h a = load_a_frag(aRowN, kt * 32, hs);
      v16h bf = load_b_frag(Wn1f, 8, kt, nt, lane);
      acc = WMMA_F16(a, bf, acc);
    }
    float bias = b_n1[nt * 16 + lrow];
    store_act_f16(acc, bias, A1[w], 128, nt, lane, true);
  }
  __syncthreads();

  // ---- n2: 128 -> 64 + residual => node_out --------------------------------
  const _Float16* aRowA = &A1[w][lrow * 128];
  for (int nt = 0; nt < 4; ++nt) {
    v8f acc = {};
#pragma unroll
    for (int kt = 0; kt < 4; ++kt) {
      v16h a = load_a_frag(aRowA, kt * 32, hs);
      v16h bf = load_b_frag(Wn2f, 4, kt, nt, lane);
      acc = WMMA_F16(a, bf, acc);
    }
    int col = nt * 16 + lrow;
    int mbase = hs * 8;
#pragma unroll
    for (int v = 0; v < 8; ++v) {
      int m = mbase + v;
      size_t node = nodeBase + m;
      float x = acc[v] + b_n2[col] + F0[node * 64 + col];
      NOh[w][m * 64 + col] = (_Float16)x;
      out1[node * 64 + col] = x;            // node_out[..., None]
    }
  }
  __syncthreads();

  // ---- gate: sigmoid(node_out @ W_g + b_g) ---------------------------------
  const _Float16* aRowO = &NOh[w][lrow * 64];
  for (int nt = 0; nt < 2; ++nt) {
    v8f acc = {};
#pragma unroll
    for (int kt = 0; kt < 2; ++kt) {
      v16h a = load_a_frag(aRowO, kt * 32, hs);
      v16h bf = load_b_frag(Wgf, 2, kt, nt, lane);
      acc = WMMA_F16(a, bf, acc);
    }
    int col = nt * 16 + lrow;
    int mbase = hs * 8;
#pragma unroll
    for (int v = 0; v < 8; ++v)
      GT[w][(mbase + v) * 32 + col] = sigmoid_f(acc[v] + b_g[col]);
  }
  __syncthreads();

  // ---- f1_out = (features1 + htype_update) * gate --------------------------
  {
    int d = lane;  // 32 lanes == d1
    for (int r = 0; r < 16; ++r) {
      size_t node = nodeBase + r;
      float g = GT[w][r * 32 + d];
      size_t base = node * 96 + d * 3;
#pragma unroll
      for (int c = 0; c < 3; ++c)
        out2[base + c] = (F1[base + c] + ht_acc[base + c]) * g;
    }
  }
}

// ---------------------------------------------------------------------------
extern "C" void kernel_launch(void* const* d_in, const int* in_sizes, int n_in,
                              void* d_out, int out_size, void* d_ws, size_t ws_size,
                              hipStream_t stream) {
  (void)in_sizes; (void)n_in; (void)out_size; (void)ws_size;

  const float* F0      = (const float*)d_in[0];   // [2,4096,64,1]
  const float* F1      = (const float*)d_in[1];   // [2,4096,32,3]
  const int*   nbrIdx  = (const int*)d_in[2];     // [2,4096,32]
  const int*   nbrMask = (const int*)d_in[3];     // [2,4096,32] (bool)
  const float* relDist = (const float*)d_in[4];   // [2,4096,32]
  const float* W_e1 = (const float*)d_in[5],  *b_e1 = (const float*)d_in[6];
  const float* W_e2 = (const float*)d_in[7],  *b_e2 = (const float*)d_in[8];
  const float* W_h1 = (const float*)d_in[9],  *b_h1 = (const float*)d_in[10];
  const float* W_h2 = (const float*)d_in[11], *b_h2 = (const float*)d_in[12];
  const float* W_n1 = (const float*)d_in[13], *b_n1 = (const float*)d_in[14];
  const float* W_n2 = (const float*)d_in[15], *b_n2 = (const float*)d_in[16];
  const float* W_g  = (const float*)d_in[17], *b_g  = (const float*)d_in[18];
  const float* ln_g = (const float*)d_in[19], *ln_b = (const float*)d_in[20];
  const float* hnS  = (const float*)d_in[21], *hnB  = (const float*)d_in[22];

  // Workspace: [m_acc 262144 f32][ht_acc 786432 f32][weight frags 109568 f16]
  float* m_acc  = (float*)d_ws;
  float* ht_acc = m_acc + 262144;
  _Float16* wf  = (_Float16*)(m_acc + 262144 + 786432);
  _Float16* W1f = wf + 0;
  _Float16* W2f = wf + 67584;
  _Float16* W3f = wf + 78848;
  _Float16* W4f = wf + 82944;
  _Float16* Wn1f = wf + 87040;
  _Float16* Wn2f = wf + 99328;
  _Float16* Wgf  = wf + 107520;

  float* out1 = (float*)d_out;          // [2,4096,64,1] = 524288 floats
  float* out2 = out1 + 524288;          // [2,4096,32,3] = 786432 floats

  prep_weights_kernel<<<dim3(132, 7), 512, 0, stream>>>(
      W_e1, W_e2, W_h1, W_h2, W_n1, W_n2, W_g, wf);
  edge_kernel<<<8192, 32, 0, stream>>>(
      F0, F1, nbrIdx, nbrMask, relDist,
      W1f, b_e1, W2f, b_e2, W3f, b_h1, W4f, b_h2,
      hnS, hnB, m_acc, ht_acc);
  node_kernel<<<512 / NW, NW * 32, 0, stream>>>(
      F0, F1, m_acc, ht_acc,
      Wn1f, b_n1, Wn2f, b_n2, Wgf, b_g, ln_g, ln_b,
      out1, out2);
}
